// CustomMHA_78194174591497
// MI455X (gfx1250) — compile-verified, matrix operand order checked by reading, not run on hardware
//
#include <hip/hip_runtime.h>
#include <hip/hip_fp16.h>

// ---------------------------------------------------------------------------
// MHA block for MI455X (gfx1250): f16 WMMA GEMMs + flash attention.
// D_MODEL=1024, N_HEADS=16, DH=64, B=4, S=2048.
// ---------------------------------------------------------------------------

typedef __attribute__((ext_vector_type(16))) _Float16 v16h;  // WMMA A/B frag
typedef __attribute__((ext_vector_type(8)))  _Float16 h8;    // 16-byte load
typedef __attribute__((ext_vector_type(8)))  float    v8f;   // WMMA C/D frag

#define D_MODEL 1024
#define N_HEADS 16
#define DH      64
#define SEQ     2048
#define BATCH   4

__device__ __forceinline__ v8f wmma_f16(v16h a, v16h b, v8f c) {
    // D = A x B + C, 16x16x32, f16 in / f32 acc
    return __builtin_amdgcn_wmma_f32_16x16x32_f16(
        /*neg_a=*/false, a, /*neg_b=*/false, b,
        /*c_mod=*/(short)0, c, /*reuse_a=*/false, /*reuse_b=*/false);
}

// Load a 16xK-contiguous f16 fragment (A layout; B layout is identical with
// lanes indexing N instead of M). Row-major source, leading dim ldk elements.
// Lanes 0-15 : row = row0+lane,    K = kb+0..7  and kb+16..23
// Lanes 16-31: row = row0+lane-16, K = kb+8..15 and kb+24..31
__device__ __forceinline__ v16h load_frag(const _Float16* __restrict__ base,
                                          int row0, int ldk, int kb, int lane) {
    const int r    = lane & 15;
    const int half = lane >> 4;
    const _Float16* p = base + (size_t)(row0 + r) * (size_t)ldk + kb + half * 8;
    h8 lo = *(const h8*)(p);
    h8 hi = *(const h8*)(p + 16);
    v16h f;
#pragma unroll
    for (int i = 0; i < 8; ++i) { f[i] = lo[i]; f[i + 8] = hi[i]; }
    return f;
}

// ---------------------------------------------------------------------------
// f32 -> f16 conversion
// ---------------------------------------------------------------------------
__global__ void cvt_f32_to_f16_kernel(const float* __restrict__ in,
                                      _Float16* __restrict__ out, int n) {
    int i = blockIdx.x * blockDim.x + threadIdx.x;
    if (i < n) out[i] = (_Float16)in[i];
}

// ---------------------------------------------------------------------------
// GEMM: C[m,n] = sum_k A[m,k] * B[n,k]   (A: MxK, B: NxK, both f16 row-major)
// Block = 256 threads = 8 waves (4 in M, 2 in N).
// Wave tile 32(M) x 64(N): 8 WMMAs per k-step from 6 fragment loads.
// Block tile 128 x 128.  M % 128 == 0, N % 128 == 0, K % 32 == 0 assumed.
// ---------------------------------------------------------------------------
template <bool OUT_F16>
__global__ void __launch_bounds__(256, 1)
gemm_wmma_f16_kernel(const _Float16* __restrict__ A,
                     const _Float16* __restrict__ B,
                     void* __restrict__ C,
                     int M, int N, int K) {
    const int lane = threadIdx.x & 31;
    const int wave = threadIdx.x >> 5;   // 0..7
    const int wm   = wave & 3;           // 0..3  (M direction)
    const int wn   = wave >> 2;          // 0..1  (N direction)
    const int mb   = blockIdx.y * 128 + wm * 32;
    const int nb   = blockIdx.x * 128 + wn * 64;

    v8f c00 = {}, c01 = {}, c02 = {}, c03 = {};
    v8f c10 = {}, c11 = {}, c12 = {}, c13 = {};

    for (int kb = 0; kb < K; kb += 32) {
        v16h a0 = load_frag(A, mb,      K, kb, lane);
        v16h a1 = load_frag(A, mb + 16, K, kb, lane);
        v16h b0 = load_frag(B, nb,      K, kb, lane);
        v16h b1 = load_frag(B, nb + 16, K, kb, lane);
        v16h b2 = load_frag(B, nb + 32, K, kb, lane);
        v16h b3 = load_frag(B, nb + 48, K, kb, lane);
        if (kb + 32 < K) {  // hint the next K-panel (global_prefetch_b8)
            __builtin_prefetch(A + (size_t)(mb + (lane & 15)) * K + kb + 32, 0, 1);
            __builtin_prefetch(B + (size_t)(nb + (lane & 15)) * K + kb + 32, 0, 1);
            __builtin_prefetch(B + (size_t)(nb + 32 + (lane & 15)) * K + kb + 32, 0, 1);
        }
        c00 = wmma_f16(a0, b0, c00);
        c01 = wmma_f16(a0, b1, c01);
        c02 = wmma_f16(a0, b2, c02);
        c03 = wmma_f16(a0, b3, c03);
        c10 = wmma_f16(a1, b0, c10);
        c11 = wmma_f16(a1, b1, c11);
        c12 = wmma_f16(a1, b2, c12);
        c13 = wmma_f16(a1, b3, c13);
    }

    const int r    = lane & 15;
    const int half = lane >> 4;
#pragma unroll
    for (int i = 0; i < 8; ++i) {
        const int m0 = mb + i + 8 * half;
        const int m1 = m0 + 16;
        const int n0 = nb + r;
        if (OUT_F16) {
            _Float16* c = (_Float16*)C;
            c[(size_t)m0 * N + n0]      = (_Float16)c00[i];
            c[(size_t)m0 * N + n0 + 16] = (_Float16)c01[i];
            c[(size_t)m0 * N + n0 + 32] = (_Float16)c02[i];
            c[(size_t)m0 * N + n0 + 48] = (_Float16)c03[i];
            c[(size_t)m1 * N + n0]      = (_Float16)c10[i];
            c[(size_t)m1 * N + n0 + 16] = (_Float16)c11[i];
            c[(size_t)m1 * N + n0 + 32] = (_Float16)c12[i];
            c[(size_t)m1 * N + n0 + 48] = (_Float16)c13[i];
        } else {
            float* c = (float*)C;
            c[(size_t)m0 * N + n0]      = c00[i];
            c[(size_t)m0 * N + n0 + 16] = c01[i];
            c[(size_t)m0 * N + n0 + 32] = c02[i];
            c[(size_t)m0 * N + n0 + 48] = c03[i];
            c[(size_t)m1 * N + n0]      = c10[i];
            c[(size_t)m1 * N + n0 + 16] = c11[i];
            c[(size_t)m1 * N + n0 + 32] = c12[i];
            c[(size_t)m1 * N + n0 + 48] = c13[i];
        }
    }
}

// ---------------------------------------------------------------------------
// RoPE + head split/transpose.
// qkv_raw: (B*S, 3*D_MODEL) f16.  Outputs:
//   q, k : (B, H, S, DH) f16 with RoPE applied
//   vT   : (B, H, DH, S) f16 (transposed so PV WMMA B-frags are K-contiguous)
// One thread per (b, s, h, d<32) rotation pair.
// ---------------------------------------------------------------------------
__global__ void rope_split_kernel(const _Float16* __restrict__ qkv_raw,
                                  _Float16* __restrict__ qo,
                                  _Float16* __restrict__ ko,
                                  _Float16* __restrict__ vTo) {
    const int tid = blockIdx.x * blockDim.x + threadIdx.x;
    const int d = tid & 31;
    const int h = (tid >> 5) & (N_HEADS - 1);
    const int s = (tid >> 9) & (SEQ - 1);
    const int b = tid >> 20;

    const size_t m    = (size_t)b * SEQ + s;
    const size_t base = m * (3 * D_MODEL) + (size_t)h * DH;

    // inv_freq = theta^(-2d/DH) = exp(-(2d/DH) * ln(10000))
    const float inv = __expf(-((float)(2 * d) / (float)DH) * 9.2103403719761836f);
    const float ang = (float)s * inv;
    float sn, cs;
    __sincosf(ang, &sn, &cs);

    const size_t qkb = ((size_t)(b * N_HEADS + h) * SEQ + s) * DH;

    // Q
    float x1 = (float)qkv_raw[base + d];
    float x2 = (float)qkv_raw[base + d + 32];
    qo[qkb + d]      = (_Float16)(x1 * cs - x2 * sn);
    qo[qkb + d + 32] = (_Float16)(x2 * cs + x1 * sn);
    // K
    x1 = (float)qkv_raw[base + D_MODEL + d];
    x2 = (float)qkv_raw[base + D_MODEL + d + 32];
    ko[qkb + d]      = (_Float16)(x1 * cs - x2 * sn);
    ko[qkb + d + 32] = (_Float16)(x2 * cs + x1 * sn);
    // V (transposed store)
    const size_t vtb = (size_t)(b * N_HEADS + h) * DH * SEQ;
    vTo[vtb + (size_t)d * SEQ + s]        = qkv_raw[base + 2 * D_MODEL + d];
    vTo[vtb + (size_t)(d + 32) * SEQ + s] = qkv_raw[base + 2 * D_MODEL + d + 32];
}

// ---------------------------------------------------------------------------
// Flash attention, causal. One wave (32 lanes) per 16-row query tile.
// Scores computed TRANSPOSED: S^T[t,s] = sum_d k[t,d] q[s,d], so in the C
// fragment each lane owns a fixed query s = lane&15 with key indices t in the
// register dimension -> softmax reduction is in-lane + one shfl_xor(16), and
// exp'd probabilities map element-for-element into the A fragment for P @ V^T.
// Only the LAST 32-key pair of each query tile needs the causal mask, so the
// interior loop is compare-free.
// ---------------------------------------------------------------------------
template <bool MASKED>
__device__ __forceinline__ void attn_process_pair(
    const _Float16* __restrict__ kbase, const _Float16* __restrict__ vbase,
    v16h qf0, v16h qf1, int tb, int lane, int sIdx,
    v8f& o0, v8f& o1, v8f& o2, v8f& o3, float& m_run, float& l_run) {
    const int half = lane >> 4;

    // K fragments (A operand, M=t over lanes, K=d)
    v16h k0a = load_frag(kbase, tb,      DH, 0,  lane);
    v16h k0b = load_frag(kbase, tb,      DH, 32, lane);
    v16h k1a = load_frag(kbase, tb + 16, DH, 0,  lane);
    v16h k1b = load_frag(kbase, tb + 16, DH, 32, lane);

    v8f s0 = {}, s1 = {};
    s0 = wmma_f16(k0a, qf0, s0);  s0 = wmma_f16(k0b, qf1, s0);
    s1 = wmma_f16(k1a, qf0, s1);  s1 = wmma_f16(k1b, qf1, s1);

    // scale 1/sqrt(DH)=0.125 (+ causal mask t<=s on the boundary pair only)
    float mx = -__builtin_inff();
#pragma unroll
    for (int i = 0; i < 8; ++i) {
        float v0 = s0[i] * 0.125f;
        float v1 = s1[i] * 0.125f;
        if (MASKED) {
            const int t0 = tb + half * 8 + i;
            if (t0 > sIdx)      v0 = -__builtin_inff();
            if (t0 + 16 > sIdx) v1 = -__builtin_inff();
        }
        s0[i] = v0; s1[i] = v1;
        mx = fmaxf(mx, fmaxf(v0, v1));
    }
    mx = fmaxf(mx, __shfl_xor(mx, 16, 32));     // other half of the t range
    const float m_new = fmaxf(m_run, mx);
    const float alpha = __expf(m_run - m_new);  // 0 on first iteration

    float sum = 0.0f;
    v16h pa;                                     // P as A fragment (f16)
#pragma unroll
    for (int i = 0; i < 8; ++i) {
        const float e0 = __expf(s0[i] - m_new);
        const float e1 = __expf(s1[i] - m_new);
        sum += e0 + e1;
        pa[i]     = (_Float16)e0;               // K = tb + klo + i
        pa[i + 8] = (_Float16)e1;               // K = tb + 16 + klo + i
    }
    sum += __shfl_xor(sum, 16, 32);
    l_run = l_run * alpha + sum;
    m_run = m_new;

    // V^T fragments (B operand, N=d over lanes, K=t)
    v16h vf0 = load_frag(vbase, 0,  SEQ, tb, lane);
    v16h vf1 = load_frag(vbase, 16, SEQ, tb, lane);
    v16h vf2 = load_frag(vbase, 32, SEQ, tb, lane);
    v16h vf3 = load_frag(vbase, 48, SEQ, tb, lane);

    // rescale accumulators per query row: O frag reg i holds s = qs+i+8*half
    float af[8];
#pragma unroll
    for (int i = 0; i < 8; ++i) af[i] = __shfl(alpha, i + 8 * half, 32);
#pragma unroll
    for (int i = 0; i < 8; ++i) {
        o0[i] *= af[i]; o1[i] *= af[i]; o2[i] *= af[i]; o3[i] *= af[i];
    }
    o0 = wmma_f16(pa, vf0, o0);
    o1 = wmma_f16(pa, vf1, o1);
    o2 = wmma_f16(pa, vf2, o2);
    o3 = wmma_f16(pa, vf3, o3);
}

__global__ void __launch_bounds__(256, 1)
attn_flash_kernel(const _Float16* __restrict__ q,
                  const _Float16* __restrict__ k,
                  const _Float16* __restrict__ vT,
                  _Float16* __restrict__ out) {
    const int lane = threadIdx.x & 31;
    const int wave = threadIdx.x >> 5;
    const int tile = blockIdx.x * 8 + wave;         // B*H*(S/16) tiles total

    const int QT = SEQ / 16;                        // 128
    const int qt = tile % QT;
    const int bh = tile / QT;                       // 0..B*H-1
    const int h  = bh & (N_HEADS - 1);
    const int b  = bh / N_HEADS;
    const int qs = qt * 16;

    const int r    = lane & 15;
    const int half = lane >> 4;
    const int sIdx = qs + r;                        // this lane's query row

    const _Float16* qbase = q  + ((size_t)bh * SEQ + qs) * DH;
    const _Float16* kbase = k  + (size_t)bh * SEQ * DH;
    const _Float16* vbase = vT + (size_t)bh * DH * SEQ;

    // Q fragments (B operand, N=s over lanes, K=d): reused across all K tiles
    const v16h qf0 = load_frag(qbase, 0, DH, 0,  lane);
    const v16h qf1 = load_frag(qbase, 0, DH, 32, lane);

    v8f o0 = {}, o1 = {}, o2 = {}, o3 = {};
    float m_run = -__builtin_inff();
    float l_run = 0.0f;

    // key range [0, qs+16): qs/32 fully-unmasked pairs + exactly 1 masked pair
    const int nfull = qs >> 5;
    for (int p = 0; p < nfull; ++p) {
        attn_process_pair<false>(kbase, vbase, qf0, qf1, p * 32, lane, sIdx,
                                 o0, o1, o2, o3, m_run, l_run);
    }
    attn_process_pair<true>(kbase, vbase, qf0, qf1, nfull * 32, lane, sIdx,
                            o0, o1, o2, o3, m_run, l_run);

    // normalize and store to (B, S, D_MODEL) layout (ready for out-projection)
    float lf[8];
#pragma unroll
    for (int i = 0; i < 8; ++i) lf[i] = __shfl(l_run, i + 8 * half, 32);
#pragma unroll
    for (int i = 0; i < 8; ++i) {
        const int srow = qs + i + 8 * half;
        const float inv = 1.0f / lf[i];
        const size_t ob = ((size_t)b * SEQ + srow) * D_MODEL + h * DH + r;
        out[ob + 0]  = (_Float16)(o0[i] * inv);
        out[ob + 16] = (_Float16)(o1[i] * inv);
        out[ob + 32] = (_Float16)(o2[i] * inv);
        out[ob + 48] = (_Float16)(o3[i] * inv);
    }
}

// ---------------------------------------------------------------------------
// Host-side orchestration
// ---------------------------------------------------------------------------
extern "C" void kernel_launch(void* const* d_in, const int* in_sizes, int n_in,
                              void* d_out, int out_size, void* d_ws, size_t ws_size,
                              hipStream_t stream) {
    const float* x   = (const float*)d_in[0];   // (4, 2048, 1024)
    const float* qkv = (const float*)d_in[1];   // (3072, 1024)
    const float* wo  = (const float*)d_in[2];   // (1024, 1024)
    float* out = (float*)d_out;                 // (4, 2048, 1024) f32

    const int M   = BATCH * SEQ;                // 8192
    const int XN  = M * D_MODEL;                // 8 Mi elems
    const int QWN = 3 * D_MODEL * D_MODEL;      // 3 Mi
    const int WON = D_MODEL * D_MODEL;          // 1 Mi
    const int RAW = M * 3 * D_MODEL;            // 24 Mi
    const int HDN = BATCH * N_HEADS * SEQ * DH; // 8 Mi (q, k, vT each)

    char* ws = (char*)d_ws;
    size_t off = 0;
    _Float16* xb    = (_Float16*)(ws + off); off += (size_t)XN  * 2;
    _Float16* qkvb  = (_Float16*)(ws + off); off += (size_t)QWN * 2;
    _Float16* wob   = (_Float16*)(ws + off); off += (size_t)WON * 2;
    _Float16* raw   = (_Float16*)(ws + off); off += (size_t)RAW * 2;
    _Float16* qbuf  = (_Float16*)(ws + off); off += (size_t)HDN * 2;
    _Float16* kbuf  = (_Float16*)(ws + off); off += (size_t)HDN * 2;
    _Float16* vtbuf = (_Float16*)(ws + off); off += (size_t)HDN * 2;
    _Float16* aout  = (_Float16*)(ws + off); off += (size_t)XN  * 2;

    // 1) precision conversion
    cvt_f32_to_f16_kernel<<<(XN  + 255) / 256, 256, 0, stream>>>(x,   xb,   XN);
    cvt_f32_to_f16_kernel<<<(QWN + 255) / 256, 256, 0, stream>>>(qkv, qkvb, QWN);
    cvt_f32_to_f16_kernel<<<(WON + 255) / 256, 256, 0, stream>>>(wo,  wob,  WON);

    // 2) QKV projection: raw = xb @ qkvb^T   (8192 x 3072 x 1024)
    {
        dim3 grid((3 * D_MODEL) / 128, M / 128);
        gemm_wmma_f16_kernel<true><<<grid, 256, 0, stream>>>(
            xb, qkvb, (void*)raw, M, 3 * D_MODEL, D_MODEL);
    }

    // 3) RoPE + split + V transpose  (B*S*H*32 threads)
    {
        const int nthreads = BATCH * SEQ * N_HEADS * 32; // 2^22
        rope_split_kernel<<<nthreads / 256, 256, 0, stream>>>(raw, qbuf, kbuf, vtbuf);
    }

    // 4) causal flash attention: one wave per 16-row query tile
    {
        const int ntiles = BATCH * N_HEADS * (SEQ / 16); // 8192
        attn_flash_kernel<<<ntiles / 8, 256, 0, stream>>>(qbuf, kbuf, vtbuf, aout);
    }

    // 5) output projection: out = aout @ wob^T  (8192 x 1024 x 1024), f32 store
    {
        dim3 grid(D_MODEL / 128, M / 128);
        gemm_wmma_f16_kernel<false><<<grid, 256, 0, stream>>>(
            aout, wob, (void*)out, M, D_MODEL, D_MODEL);
    }
}
